// task2model_29996051595843
// MI455X (gfx1250) — matrix operand order repaired and checked by reading, not compiled, not run on hardware
//
#include <hip/hip_runtime.h>
#include <math.h>

// ---------------- model constants ----------------
#define BATCH   256
#define HH      64
#define WWID    100
#define HWSZ    (HH*WWID)          // 6400
#define CODE_SZ 800
#define FLATSZ  12800              // 2*64*100
#define NBITS   64
#define KP_DEC  832                // CODE_SZ zero-padded up to a multiple of 64

typedef __attribute__((ext_vector_type(16))) _Float16 v16h;
typedef __attribute__((ext_vector_type(8)))  _Float16 v8h;
typedef __attribute__((ext_vector_type(8)))  float    v8f;

__device__ __forceinline__ float actf(float x, int act) {
    switch (act) {
        case 1:  return x > 0.f ? x : 0.3f * x;                // leaky relu 0.3
        case 2:  return 1.f / (1.f + __expf(-x));              // sigmoid
        case 3:  return tanhf(x);                              // tanh
        default: return x;
    }
}

// ---------------- preprocess: stack + normalize ----------------
__global__ void k_prep(const float* __restrict__ hr, const float* __restrict__ hi,
                       float* __restrict__ x, int n /* BATCH*HWSZ */) {
    int i = blockIdx.x * blockDim.x + threadIdx.x;
    if (i >= n) return;
    int b = i / HWSZ, p = i % HWSZ;
    x[((size_t)b * 2 + 0) * HWSZ + p] = (hr[i] - 0.1f)  * (1.f / 1.2f);
    x[((size_t)b * 2 + 1) * HWSZ + p] = (hi[i] + 0.05f) * (1.f / 0.9f);
}

// ---------------- direct conv + BN + act (+ fused residual relu, optional f16 copy) ----
__global__ void k_conv_bn(const float* __restrict__ in, const float* __restrict__ w,
                          const float* __restrict__ g,  const float* __restrict__ bb,
                          const float* __restrict__ mm, const float* __restrict__ vv,
                          const float* __restrict__ res, float* __restrict__ out,
                          _Float16* __restrict__ out16,
                          int Cin, int Cout, int Kk, int act, int fuse_res, int n) {
    int i = blockIdx.x * blockDim.x + threadIdx.x;
    if (i >= n) return;
    int xx = i % WWID;
    int yy = (i / WWID) % HH;
    int co = (i / HWSZ) % Cout;
    int b  = i / (HWSZ * Cout);
    int pad = (Kk - 1) >> 1;
    float s = 0.f;
    const float* wc = w + (size_t)co * Cin * Kk * Kk;
    for (int ci = 0; ci < Cin; ++ci) {
        const float* ip = in + ((size_t)b * Cin + ci) * HWSZ;
        const float* wk = wc + (size_t)ci * Kk * Kk;
        for (int ky = 0; ky < Kk; ++ky) {
            int iy = yy + ky - pad;
            if ((unsigned)iy >= (unsigned)HH) continue;
            const float* iprow = ip + iy * WWID;
            const float* wkrow = wk + ky * Kk;
            for (int kx = 0; kx < Kk; ++kx) {
                int ix = xx + kx - pad;
                if ((unsigned)ix >= (unsigned)WWID) continue;
                s = fmaf(iprow[ix], wkrow[kx], s);
            }
        }
    }
    float scale = g[co] * rsqrtf(vv[co] + 1e-5f);
    float val   = fmaf(s, scale, bb[co] - mm[co] * scale);
    val = actf(val, act);
    if (fuse_res) val = fmaxf(val + res[i], 0.f);   // d = relu(tanh(conv) + d)
    out[i] = val;
    if (out16) out16[i] = (_Float16)val;
}

// ---------------- f32 -> f16 convert with K padding (rows x K -> rows x KP) --------
__global__ void k_f16_pad(const float* __restrict__ in, _Float16* __restrict__ out,
                          int K, int KP, int n /* rows*KP */) {
    int i = blockIdx.x * blockDim.x + threadIdx.x;
    if (i >= n) return;
    int r = i / KP, c = i % KP;
    out[i] = (c < K) ? (_Float16)in[(size_t)r * K + c] : (_Float16)0.f;
}

// ---------------- fragment loader for the WMMA GEMM ----------------
// A-frag (contiguous k order used by this code, matched by B): lane (hi=l>>4, m=l&15) holds
//   halves[0..7]  = A[m, k0 + hi*16 .. +7]
//   halves[8..15] = A[m, k0 + hi*16 + 8 .. +7]      (two contiguous 16B loads)
// B-frag (same contiguous k order as A): lane holds column n=l&15, K = k0 + (l>>4)*16 .. +15,
//   i.e. 16 contiguous halves of W row n (two 16B loads).
__device__ __forceinline__ void load_frags(const _Float16* __restrict__ arow,
                                           const _Float16* const* wrow,
                                           int k0, int hi,
                                           v16h& a, v16h* b) {
    v8h alo = *(const v8h*)(arow + k0 + hi * 16);
    v8h ahi = *(const v8h*)(arow + k0 + hi * 16 + 8);
#pragma unroll
    for (int j = 0; j < 8; ++j) { a[j] = alo[j]; a[8 + j] = ahi[j]; }
#pragma unroll
    for (int t = 0; t < 4; ++t) {
        const _Float16* wp = wrow[t] + k0 + hi * 16;
        v8h b0 = *(const v8h*)wp;
        v8h b1 = *(const v8h*)(wp + 8);
#pragma unroll
        for (int j = 0; j < 8; ++j) { b[t][j] = b0[j]; b[t][8 + j] = b1[j]; }
    }
}

// ---------------- WMMA GEMM: C[M,N] = act(A[M,KP] * W[N,KP]^T + bias) --------------
// One wave per 16(M) x 64(N) tile; KP % 64 == 0 (zero-padded K).
// Manual 2x-unrolled ping-pong: two named fragment sets, no rotation copies.
__global__ __launch_bounds__(32)
void k_gemm_wmma(const _Float16* __restrict__ A, const _Float16* __restrict__ Wm,
                 const float* __restrict__ bias, float* __restrict__ C,
                 int M, int N, int KP, int act) {
    const int lane = threadIdx.x & 31;
    const int m0   = blockIdx.x * 16;
    const int n0   = blockIdx.y * 64;
    const int mrow = lane & 15;
    const int hi   = lane >> 4;

    v8f acc[4] = {v8f{}, v8f{}, v8f{}, v8f{}};

    const _Float16* arow = A + (size_t)(m0 + mrow) * KP;
    int nIdx[4];
    const _Float16* wrow[4];
#pragma unroll
    for (int t = 0; t < 4; ++t) {
        int n = n0 + t * 16 + mrow;
        nIdx[t] = n;
        if (n >= N) n = N - 1;                 // clamp (store guarded later)
        wrow[t] = Wm + (size_t)n * KP;
    }

    v16h a0, b0[4], a1, b1[4];
    load_frags(arow, wrow, 0, hi, a0, b0);     // prologue

    int k = 0;
    for (; k + 64 < KP; k += 64) {
        __builtin_prefetch(arow + k + 96, 0, 3);
        __builtin_prefetch(wrow[0] + k + 96, 0, 3);
        __builtin_prefetch(wrow[2] + k + 96, 0, 3);
        load_frags(arow, wrow, k + 32, hi, a1, b1);
#pragma unroll
        for (int t = 0; t < 4; ++t)
            acc[t] = __builtin_amdgcn_wmma_f32_16x16x32_f16(
                false, a0, false, b0[t], (short)0, acc[t], false, false);
        load_frags(arow, wrow, k + 64, hi, a0, b0);
#pragma unroll
        for (int t = 0; t < 4; ++t)
            acc[t] = __builtin_amdgcn_wmma_f32_16x16x32_f16(
                false, a1, false, b1[t], (short)0, acc[t], false, false);
    }
    // tail: exactly 64 K-values remain (KP % 64 == 0)
    load_frags(arow, wrow, k + 32, hi, a1, b1);
#pragma unroll
    for (int t = 0; t < 4; ++t)
        acc[t] = __builtin_amdgcn_wmma_f32_16x16x32_f16(
            false, a0, false, b0[t], (short)0, acc[t], false, false);
#pragma unroll
    for (int t = 0; t < 4; ++t)
        acc[t] = __builtin_amdgcn_wmma_f32_16x16x32_f16(
            false, a1, false, b1[t], (short)0, acc[t], false, false);

    // C/D layout: VGPR r, lanes 0-15 -> M=r,N=lane; lanes 16-31 -> M=r+8,N=lane-16
#pragma unroll
    for (int t = 0; t < 4; ++t) {
        int n = nIdx[t];
        if (n >= N) continue;
        float bv = bias ? bias[n] : 0.f;
#pragma unroll
        for (int r = 0; r < 8; ++r) {
            int m = m0 + hi * 8 + r;
            float val = acc[t][r] + bv;
            val = actf(val, act);
            C[(size_t)m * N + n] = val;
        }
    }
}

// ------- 64-bit binary-fraction quantizer (f16 out, K-padded rows for dec GEMM) ----
__global__ void k_quantize(const float* __restrict__ in, _Float16* __restrict__ out,
                           int K, int KP, int n /* rows*KP */) {
    int i = blockIdx.x * blockDim.x + threadIdx.x;
    if (i >= n) return;
    int r = i / KP, c = i % KP;
    if (c >= K) { out[i] = (_Float16)0.f; return; }
    float d = in[(size_t)r * K + c];
    float acc = 0.f, wq = 0.5f;
    for (int k = 0; k < NBITS; ++k) {
        d = d * 2.f;
        float bit = (d >= 1.f) ? 1.f : 0.f;
        d -= bit;
        acc = fmaf(bit, wq, acc);
        wq *= 0.5f;
    }
    out[i] = (_Float16)acc;
}

// ---------------- final sigmoid + channel rescale ----------------
__global__ void k_final(const float* __restrict__ d, float* __restrict__ out, int n) {
    int i = blockIdx.x * blockDim.x + threadIdx.x;
    if (i >= n) return;
    int c = (i / HWSZ) % 2;
    float s = 1.f / (1.f + __expf(-d[i]));
    if (c == 0) { s = fmaf(s, 1.2f, 0.1f); s = fmaf(s, 0.9f, -0.05f); }  // (s*STD_R+MEAN_R)*STD_I+MEAN_I
    out[i] = s;
}

// ---------------- host ----------------
struct ConvP { const float *w, *g, *b, *m, *v; };
static ConvP getcp(void* const* d_in, int& p) {
    ConvP c{ (const float*)d_in[p],     (const float*)d_in[p + 1],
             (const float*)d_in[p + 2], (const float*)d_in[p + 3],
             (const float*)d_in[p + 4] };
    p += 5;
    return c;
}

extern "C" void kernel_launch(void* const* d_in, const int* in_sizes, int n_in,
                              void* d_out, int out_size, void* d_ws, size_t ws_size,
                              hipStream_t stream) {
    (void)in_sizes; (void)n_in; (void)out_size; (void)ws_size;

    const float* Hr = (const float*)d_in[0];
    const float* Hi = (const float*)d_in[1];
    int p = 2;
    ConvP enc1 = getcp(d_in, p);                 // 2..6
    ConvP enc2 = getcp(d_in, p);                 // 7..11
    const float* enc_fc_w = (const float*)d_in[p++];   // (800,12800)
    const float* enc_fc_b = (const float*)d_in[p++];   // (800,)
    const float* dec_fc_w = (const float*)d_in[p++];   // (12800,800)
    const float* dec_fc_b = (const float*)d_in[p++];   // (12800,)
    ConvP decc = getcp(d_in, p);                 // 16..20
    ConvP rc[5][3];
    for (int i = 0; i < 5; ++i)
        for (int j = 0; j < 3; ++j)
            rc[i][j] = getcp(d_in, p);
    // d_in[p] = is_training (ignored; inference path = quantize)

    // workspace carving
    char* ws = (char*)d_ws;
    size_t off = 0;
    auto alloc = [&](size_t bytes) -> void* {
        void* r = ws + off;
        off = (off + bytes + 255) & ~(size_t)255;
        return r;
    };
    const size_t n2  = (size_t)BATCH * 2  * HWSZ;   // 3,276,800
    const size_t n8  = (size_t)BATCH * 8  * HWSZ;
    const size_t n16 = (size_t)BATCH * 16 * HWSZ;
    const size_t w16elems = (size_t)FLATSZ * KP_DEC;   // max(800*12800, 12800*832)
    float*    bufA  = (float*)alloc(n2 * 4);
    float*    bufB  = (float*)alloc(n2 * 4);
    float*    bufC  = (float*)alloc(n2 * 4);
    float*    buf8  = (float*)alloc(n8 * 4);
    float*    buf16 = (float*)alloc(n16 * 4);
    _Float16* A16   = (_Float16*)alloc((size_t)BATCH * FLATSZ * 2);
    _Float16* W16   = (_Float16*)alloc(w16elems * 2);              // reused enc/dec
    float*    henc  = (float*)alloc((size_t)BATCH * CODE_SZ * 4);
    _Float16* H16   = (_Float16*)alloc((size_t)BATCH * KP_DEC * 2);

    const int TB = 256;
    auto g1 = [&](size_t n) { return dim3((unsigned)((n + TB - 1) / TB)); };

    // encoder
    size_t nPix = (size_t)BATCH * HWSZ;
    k_prep<<<g1(nPix), TB, 0, stream>>>(Hr, Hi, bufA, (int)nPix);
    k_conv_bn<<<g1(n2), TB, 0, stream>>>(bufA, enc1.w, enc1.g, enc1.b, enc1.m, enc1.v,
                                         nullptr, bufB, nullptr,
                                         2, 2, 7, /*lrelu*/1, 0, (int)n2);
    // enc conv2 writes the f16 GEMM operand directly (row stride == FLATSZ == KP)
    k_conv_bn<<<g1(n2), TB, 0, stream>>>(bufB, enc2.w, enc2.g, enc2.b, enc2.m, enc2.v,
                                         nullptr, bufA, A16,
                                         2, 2, 7, /*lrelu*/1, 0, (int)n2);
    // enc FC (WMMA, fused sigmoid), K = KP = 12800
    k_f16_pad<<<g1((size_t)CODE_SZ * FLATSZ), TB, 0, stream>>>(enc_fc_w, W16,
                                                               FLATSZ, FLATSZ,
                                                               CODE_SZ * FLATSZ);
    dim3 ge(BATCH / 16, (CODE_SZ + 63) / 64);
    k_gemm_wmma<<<ge, 32, 0, stream>>>(A16, W16, enc_fc_b, henc,
                                       BATCH, CODE_SZ, FLATSZ, /*sigmoid*/2);
    // quantize -> f16 dec-GEMM operand, K=800 padded to KP=832
    k_quantize<<<g1((size_t)BATCH * KP_DEC), TB, 0, stream>>>(henc, H16,
                                                              CODE_SZ, KP_DEC,
                                                              BATCH * KP_DEC);
    // dec FC (WMMA), K=800 padded to KP=832
    k_f16_pad<<<g1((size_t)FLATSZ * KP_DEC), TB, 0, stream>>>(dec_fc_w, W16,
                                                              CODE_SZ, KP_DEC,
                                                              FLATSZ * KP_DEC);
    dim3 gd(BATCH / 16, FLATSZ / 64);
    k_gemm_wmma<<<gd, 32, 0, stream>>>(H16, W16, dec_fc_b, bufC,
                                       BATCH, FLATSZ, KP_DEC, /*none*/0);
    // dec conv (sigmoid)
    k_conv_bn<<<g1(n2), TB, 0, stream>>>(bufC, decc.w, decc.g, decc.b, decc.m, decc.v,
                                         nullptr, bufA, nullptr,
                                         2, 2, 7, /*sigmoid*/2, 0, (int)n2);
    // refine blocks
    float* dcur = bufA;
    float* dnxt = bufB;
    for (int i = 0; i < 5; ++i) {
        k_conv_bn<<<g1(n8), TB, 0, stream>>>(dcur, rc[i][0].w, rc[i][0].g, rc[i][0].b,
                                             rc[i][0].m, rc[i][0].v, nullptr, buf8, nullptr,
                                             2, 8, 7, /*lrelu*/1, 0, (int)n8);
        k_conv_bn<<<g1(n16), TB, 0, stream>>>(buf8, rc[i][1].w, rc[i][1].g, rc[i][1].b,
                                              rc[i][1].m, rc[i][1].v, nullptr, buf16, nullptr,
                                              8, 16, 5, /*lrelu*/1, 0, (int)n16);
        k_conv_bn<<<g1(n2), TB, 0, stream>>>(buf16, rc[i][2].w, rc[i][2].g, rc[i][2].b,
                                             rc[i][2].m, rc[i][2].v, /*res=*/dcur, dnxt, nullptr,
                                             16, 2, 3, /*tanh*/3, /*fuse_res*/1, (int)n2);
        float* t = dcur; dcur = dnxt; dnxt = t;
    }
    // final sigmoid + rescale
    k_final<<<g1(n2), TB, 0, stream>>>(dcur, (float*)d_out, (int)n2);
}
